// MultiHeadAttention_2697239461944
// MI455X (gfx1250) — matrix operand order repaired and checked by reading, not compiled, not run on hardware
//
#include <hip/hip_runtime.h>
#include <hip/hip_bf16.h>
#include <math.h>
#include <stdint.h>

#define B_  2
#define S_  4096
#define D_  512
#define H_  8
#define DK_ 64
#define MROWS (B_ * S_)   // 8192 rows for the projection GEMMs

typedef __attribute__((ext_vector_type(16))) __bf16 v16bf;
typedef __attribute__((ext_vector_type(8)))  float  v8f;
typedef __attribute__((ext_vector_type(4)))  int    v4i;

#if defined(__has_builtin) && __has_builtin(__builtin_amdgcn_global_load_async_to_lds_b128)
#define HAVE_ASYNC_LD 1
#else
#define HAVE_ASYNC_LD 0
#endif
#if defined(__has_builtin) && __has_builtin(__builtin_amdgcn_global_store_async_from_lds_b128)
#define HAVE_ASYNC_ST 1
#else
#define HAVE_ASYNC_ST 0
#endif

__device__ __forceinline__ void wait_async0() {
#if defined(__has_builtin) && __has_builtin(__builtin_amdgcn_s_wait_asynccnt)
  __builtin_amdgcn_s_wait_asynccnt(0);
#else
  asm volatile("s_wait_asynccnt 0" ::: "memory");
#endif
}

// 16B global -> LDS via GLOBAL_LOAD_ASYNC_TO_LDS_B128 (ASYNCcnt).
__device__ __forceinline__ void stage16(const unsigned short* g, unsigned short* l) {
#if HAVE_ASYNC_LD
  using gptr_t = __attribute__((address_space(1))) v4i*;
  using lptr_t = __attribute__((address_space(3))) v4i*;
  __builtin_amdgcn_global_load_async_to_lds_b128(
      (gptr_t)(unsigned long long)(uintptr_t)g,
      (lptr_t)(unsigned int)(uintptr_t)l, 0, 0);
#else
  *(uint4*)l = *(const uint4*)g;
#endif
}

// 16B LDS -> global via GLOBAL_STORE_ASYNC_FROM_LDS_B128 (ASYNCcnt).
// cpol = 1 -> TH_STORE_NT: attn is a 1 GB write-once stream; keep it out of L2.
__device__ __forceinline__ void store16_from_lds(float* g, const float* l) {
#if HAVE_ASYNC_ST
  using gptr_t = __attribute__((address_space(1))) v4i*;
  using lptr_t = __attribute__((address_space(3))) v4i*;
  __builtin_amdgcn_global_store_async_from_lds_b128(
      (gptr_t)(unsigned long long)(uintptr_t)g,
      (lptr_t)(unsigned int)(uintptr_t)l, 0, 1);
#else
  *(float4*)g = *(const float4*)l;
#endif
}

__device__ __forceinline__ void stage_fence() {
#if HAVE_ASYNC_LD || HAVE_ASYNC_ST
  wait_async0();
#endif
}

__device__ __forceinline__ unsigned short f2bf(float f) {
  __bf16 h = (__bf16)f;                 // native cvt; fuses into v_fma_mix*_bf16
  unsigned short u;
  __builtin_memcpy(&u, &h, 2);
  return u;
}

__device__ __forceinline__ v8f v8f_zero() {
  v8f z;
#pragma unroll
  for (int i = 0; i < 8; ++i) z[i] = 0.0f;
  return z;
}

union BF16FragQ { v16bf v; uint4 q[2]; };

// A-matrix fragment (16x32 bf16), LDS [row][k]; two contiguous 16B runs
// (K=0..7|+8*half and K=16..23|+8*half) -> 2x ds_load_b128 per fragment.
__device__ __forceinline__ v16bf load_a_frag(const unsigned short* lds, int stride,
                                             int mBase, int kBase, int lane) {
  const int l15 = lane & 15, half = lane >> 4;
  const unsigned short* row = lds + (size_t)(mBase + l15) * stride + kBase + 8 * half;
  BF16FragQ f;
  f.q[0] = *(const uint4*)(row);        // K = 0..7   (+8*half)
  f.q[1] = *(const uint4*)(row + 16);   // K = 16..23 (+8*half)
  return f.v;
}

// B-matrix fragment (32x16 bf16), LDS TRANSPOSED [n][k]; two contiguous 16B runs
// (K=0..7|+8*half and K=16..23|+8*half) -> 2x ds_load_b128 per fragment.
__device__ __forceinline__ v16bf load_b_frag(const unsigned short* lds, int stride,
                                             int nBase, int kBase, int lane) {
  const int l15 = lane & 15, half = lane >> 4;
  const unsigned short* row = lds + (size_t)(nBase + l15) * stride + kBase + 8 * half;
  BF16FragQ f;
  f.q[0] = *(const uint4*)(row);
  f.q[1] = *(const uint4*)(row + 16);
  return f.v;
}

__device__ __forceinline__ v8f wmma_bf16(v16bf a, v16bf b, v8f c) {
  return __builtin_amdgcn_wmma_f32_16x16x32_bf16(false, a, false, b, (short)0, c,
                                                 false, false);
}

// ---------------------------------------------------------------------------
// GEMM 1: Y_bf16[B,H,S,DK] = (X_f32[M,512] @ W + bias) * outScale  (head-split)
// ---------------------------------------------------------------------------
__global__ __launch_bounds__(128)
void proj_f32_bf16hs_kernel(const float* __restrict__ X, const float* __restrict__ W,
                            const float* __restrict__ bias, unsigned short* __restrict__ Y,
                            float outScale) {
  __shared__ unsigned short lx[64][40];    // 80B rows: 16B-aligned, 20-bank stride
  __shared__ unsigned short lwT[64][40];   // W tile transposed [n][k]

  const int tid = threadIdx.x;
  const int lane = tid & 31, wave = tid >> 5;
  const int l15 = lane & 15, half = lane >> 4;
  const int n0 = blockIdx.x * 64;
  const int m0 = blockIdx.y * 64;

  v8f acc[4];
#pragma unroll
  for (int i = 0; i < 4; ++i) acc[i] = v8f_zero();

  for (int k0 = 0; k0 < D_; k0 += 32) {
    __syncthreads();
#pragma unroll
    for (int i = 0; i < 16; ++i) {                 // 64x32 X tile, f32->bf16
      int idx = tid + i * 128;
      int r = idx >> 5, c = idx & 31;
      lx[r][c] = f2bf(X[(size_t)(m0 + r) * D_ + k0 + c]);
    }
#pragma unroll
    for (int i = 0; i < 16; ++i) {                 // 32x64 W tile -> lwT[n][k]
      int idx = tid + i * 128;
      int r = idx >> 6, c = idx & 63;              // r = k, c = n
      lwT[c][r] = f2bf(W[(size_t)(k0 + r) * D_ + n0 + c]);
    }
    __syncthreads();

    v16bf a = load_a_frag(&lx[0][0], 40, wave * 16, 0, lane);
#pragma unroll
    for (int nt = 0; nt < 4; ++nt) {
      v16bf b = load_b_frag(&lwT[0][0], 40, nt * 16, 0, lane);
      acc[nt] = wmma_bf16(a, b, acc[nt]);
    }
  }

#pragma unroll
  for (int nt = 0; nt < 4; ++nt) {
    int n = n0 + nt * 16 + l15;
    float bv = bias[n];
#pragma unroll
    for (int r = 0; r < 8; ++r) {
      int m = m0 + wave * 16 + r + 8 * half;
      float v = (acc[nt][r] + bv) * outScale;
      int bb = m >> 12, s = m & (S_ - 1);
      int h = n >> 6, dk = n & 63;
      Y[((size_t)(bb * H_ + h) * S_ + s) * DK_ + dk] = f2bf(v);
    }
  }
}

// ---------------------------------------------------------------------------
// GEMM 2: out_f32[M,512] = Xb_bf16[M,512] @ W + bias   (final projection)
// ---------------------------------------------------------------------------
__global__ __launch_bounds__(128)
void proj_bf16_f32_kernel(const unsigned short* __restrict__ Xb,
                          const float* __restrict__ W,
                          const float* __restrict__ bias, float* __restrict__ Y) {
  __shared__ unsigned short lx[64][40];
  __shared__ unsigned short lwT[64][40];

  const int tid = threadIdx.x;
  const int lane = tid & 31, wave = tid >> 5;
  const int l15 = lane & 15, half = lane >> 4;
  const int n0 = blockIdx.x * 64;
  const int m0 = blockIdx.y * 64;

  v8f acc[4];
#pragma unroll
  for (int i = 0; i < 4; ++i) acc[i] = v8f_zero();

  for (int k0 = 0; k0 < D_; k0 += 32) {
    __syncthreads();
#pragma unroll
    for (int i = 0; i < 2; ++i) {                  // 64x32 bf16 tile: 256 x b128 async
      int idx = tid + i * 128;
      int r = idx >> 2, c8 = (idx & 3) * 8;
      stage16(Xb + (size_t)(m0 + r) * D_ + k0 + c8, &lx[r][c8]);
    }
#pragma unroll
    for (int i = 0; i < 16; ++i) {                 // weights f32 -> bf16, transposed
      int idx = tid + i * 128;
      int r = idx >> 6, c = idx & 63;
      lwT[c][r] = f2bf(W[(size_t)(k0 + r) * D_ + n0 + c]);
    }
    stage_fence();
    __syncthreads();

    v16bf a = load_a_frag(&lx[0][0], 40, wave * 16, 0, lane);
#pragma unroll
    for (int nt = 0; nt < 4; ++nt) {
      v16bf b = load_b_frag(&lwT[0][0], 40, nt * 16, 0, lane);
      acc[nt] = wmma_bf16(a, b, acc[nt]);
    }
  }

#pragma unroll
  for (int nt = 0; nt < 4; ++nt) {
    int n = n0 + nt * 16 + l15;
    float bv = bias[n];
#pragma unroll
    for (int r = 0; r < 8; ++r) {
      int m = m0 + wave * 16 + r + 8 * half;
      Y[(size_t)m * D_ + n] = acc[nt][r] + bv;
    }
  }
}

// ---------------------------------------------------------------------------
// Attention per (b, h, 64-row q block). q/k/v bf16 head-split [B,H,S,DK];
// q pre-scaled by 1/sqrt(DK). Two-pass causal softmax; attn tile emitted via
// async NT b128 stores from LDS; ctx accumulated via WMMA, stored bf16 [B,S,D].
// ---------------------------------------------------------------------------
__global__ __launch_bounds__(128)
void attention_kernel(const unsigned short* __restrict__ qp,
                      const unsigned short* __restrict__ kp,
                      const unsigned short* __restrict__ vp,
                      float* __restrict__ attn,
                      unsigned short* __restrict__ ctx) {
  __shared__ unsigned short sq[64][72];    // Q tile [m][dk]  (144B rows)
  __shared__ unsigned short skT[64][72];   // K tile [keyIdx][dk]
  __shared__ unsigned short svT[64][72];   // V tile transposed [dk][keyIdx]
  __shared__ unsigned short sp[64][72];    // P tile [m][keyIdx] bf16 (WMMA A)
  __shared__ float          spf[64][64];   // P tile fp32 (async store source)

  const int tid = threadIdx.x;
  const int lane = tid & 31, wave = tid >> 5;
  const int l15 = lane & 15, half = lane >> 4;
  const int qb = blockIdx.x, h = blockIdx.y, b = blockIdx.z;
  const size_t headBase = (size_t)(b * H_ + h) * S_ * DK_;
  const size_t attnBase = (size_t)(b * H_ + h) * S_ * S_;
  const int rowBase = qb * 64;

#pragma unroll
  for (int i = 0; i < 4; ++i) {            // stage Q: 512 x b128 async
    int idx = tid + i * 128;
    int r = idx >> 3, c8 = (idx & 7) * 8;
    stage16(qp + headBase + (size_t)(rowBase + r) * DK_ + c8, &sq[r][c8]);
  }
  stage_fence();
  __syncthreads();

  float mrow[8], lrow[8];
#pragma unroll
  for (int r = 0; r < 8; ++r) { mrow[r] = -3.0e38f; lrow[r] = 0.0f; }

  // ---------------- pass 1: row max / sum of exp ----------------
  for (int kb = 0; kb <= qb; ++kb) {
    __syncthreads();
#pragma unroll
    for (int i = 0; i < 4; ++i) {          // stage K tile: 512 x b128 async
      int idx = tid + i * 128;
      int r = idx >> 3, c8 = (idx & 7) * 8;
      stage16(kp + headBase + (size_t)(kb * 64 + r) * DK_ + c8, &skT[r][c8]);
    }
    stage_fence();
    __syncthreads();

    v8f s[4];
#pragma unroll
    for (int i = 0; i < 4; ++i) s[i] = v8f_zero();
#pragma unroll
    for (int ks = 0; ks < 2; ++ks) {
      v16bf a = load_a_frag(&sq[0][0], 72, wave * 16, ks * 32, lane);
#pragma unroll
      for (int nt = 0; nt < 4; ++nt) {
        v16bf bb = load_b_frag(&skT[0][0], 72, nt * 16, ks * 32, lane);
        s[nt] = wmma_bf16(a, bb, s[nt]);
      }
    }
    if (kb == qb) {                        // causal mask on diagonal block
#pragma unroll
      for (int nt = 0; nt < 4; ++nt) {
        int col = kb * 64 + nt * 16 + l15;
#pragma unroll
        for (int r = 0; r < 8; ++r) {
          int row = rowBase + wave * 16 + r + 8 * half;
          if (col > row) s[nt][r] = -1.0e9f;
        }
      }
    }
#pragma unroll
    for (int r = 0; r < 8; ++r) {          // per-row stats; xor-shfl stays in half
      float tm = fmaxf(fmaxf(s[0][r], s[1][r]), fmaxf(s[2][r], s[3][r]));
#pragma unroll
      for (int m = 1; m <= 8; m <<= 1) tm = fmaxf(tm, __shfl_xor(tm, m, 32));
      float mnew = fmaxf(mrow[r], tm);
      float ps = __expf(s[0][r] - mnew) + __expf(s[1][r] - mnew)
               + __expf(s[2][r] - mnew) + __expf(s[3][r] - mnew);
#pragma unroll
      for (int m = 1; m <= 8; m <<= 1) ps += __shfl_xor(ps, m, 32);
      lrow[r] = lrow[r] * __expf(mrow[r] - mnew) + ps;
      mrow[r] = mnew;
    }
  }

  float linv[8];
#pragma unroll
  for (int r = 0; r < 8; ++r) linv[r] = 1.0f / lrow[r];

  v8f cacc[4];
#pragma unroll
  for (int i = 0; i < 4; ++i) cacc[i] = v8f_zero();

  // ---------------- pass 2: write attn once (async NT), ctx += P@V ----------------
  const int nkb = S_ / 64;
  for (int kb = 0; kb < nkb; ++kb) {
    if (kb > qb) {                         // fully masked block -> zeros
      float4 z = make_float4(0.f, 0.f, 0.f, 0.f);
#pragma unroll
      for (int i = 0; i < 8; ++i) {
        int idx = tid + i * 128;
        int r = idx >> 4, c4 = idx & 15;
        *(float4*)(attn + attnBase + (size_t)(rowBase + r) * S_ + kb * 64 + c4 * 4) = z;
      }
      continue;
    }
    __syncthreads();
#pragma unroll
    for (int i = 0; i < 4; ++i) {          // stage K tile (again): b128 async
      int idx = tid + i * 128;
      int r = idx >> 3, c8 = (idx & 7) * 8;
      stage16(kp + headBase + (size_t)(kb * 64 + r) * DK_ + c8, &skT[r][c8]);
    }
#pragma unroll
    for (int i = 0; i < 16; ++i) {         // stage V transposed: [dk][keyIdx]
      int idx = tid + i * 128;
      int r = idx >> 5, c2 = (idx & 31) * 2;
      unsigned int v2 = *(const unsigned int*)(vp + headBase +
                                               (size_t)(kb * 64 + r) * DK_ + c2);
      svT[c2][r]     = (unsigned short)(v2 & 0xFFFFu);
      svT[c2 + 1][r] = (unsigned short)(v2 >> 16);
    }
    stage_fence();                         // also retires prior async attn stores
    __syncthreads();

    v8f s[4];
#pragma unroll
    for (int i = 0; i < 4; ++i) s[i] = v8f_zero();
#pragma unroll
    for (int ks = 0; ks < 2; ++ks) {
      v16bf a = load_a_frag(&sq[0][0], 72, wave * 16, ks * 32, lane);
#pragma unroll
      for (int nt = 0; nt < 4; ++nt) {
        v16bf bb = load_b_frag(&skT[0][0], 72, nt * 16, ks * 32, lane);
        s[nt] = wmma_bf16(a, bb, s[nt]);
      }
    }
    if (kb == qb) {
#pragma unroll
      for (int nt = 0; nt < 4; ++nt) {
        int col = kb * 64 + nt * 16 + l15;
#pragma unroll
        for (int r = 0; r < 8; ++r) {
          int row = rowBase + wave * 16 + r + 8 * half;
          if (col > row) s[nt][r] = -1.0e9f;
        }
      }
    }
#pragma unroll
    for (int nt = 0; nt < 4; ++nt) {       // normalize; stash fp32 + bf16 in LDS
      int colL = nt * 16 + l15;
#pragma unroll
      for (int r = 0; r < 8; ++r) {
        int rowL = wave * 16 + r + 8 * half;
        float p = __expf(s[nt][r] - mrow[r]) * linv[r];
        spf[rowL][colL] = p;
        sp[rowL][colL] = f2bf(p);
      }
    }
    __syncthreads();
#pragma unroll
    for (int i = 0; i < 8; ++i) {          // attn tile: 1024 x async NT b128
      int idx = tid + i * 128;
      int r = idx >> 4, c4 = (idx & 15) * 4;
      store16_from_lds(attn + attnBase + (size_t)(rowBase + r) * S_ + kb * 64 + c4,
                       &spf[r][c4]);
    }
#pragma unroll
    for (int ks = 0; ks < 2; ++ks) {       // ctx += P @ V (overlaps store drain)
      v16bf a = load_a_frag(&sp[0][0], 72, wave * 16, ks * 32, lane);
#pragma unroll
      for (int dt = 0; dt < 4; ++dt) {
        v16bf bb = load_b_frag(&svT[0][0], 72, dt * 16, ks * 32, lane);
        cacc[dt] = wmma_bf16(a, bb, cacc[dt]);
      }
    }
  }

#pragma unroll
  for (int dt = 0; dt < 4; ++dt) {         // ctx -> bf16 [B,S,D] merged-head
    int dk = dt * 16 + l15;
#pragma unroll
    for (int r = 0; r < 8; ++r) {
      int rowL = wave * 16 + r + 8 * half;
      ctx[((size_t)b * S_ + rowBase + rowL) * D_ + h * DK_ + dk] = f2bf(cacc[dt][r]);
    }
  }
}

// ---------------------------------------------------------------------------
extern "C" void kernel_launch(void* const* d_in, const int* in_sizes, int n_in,
                              void* d_out, int out_size, void* d_ws, size_t ws_size,
                              hipStream_t stream) {
  const float* Q  = (const float*)d_in[0];
  const float* K  = (const float*)d_in[1];
  const float* V  = (const float*)d_in[2];
  const float* wq = (const float*)d_in[3];
  const float* bq = (const float*)d_in[4];
  const float* wk = (const float*)d_in[5];
  const float* bk = (const float*)d_in[6];
  const float* wv = (const float*)d_in[7];
  const float* bv = (const float*)d_in[8];
  const float* wo = (const float*)d_in[9];
  const float* bo = (const float*)d_in[10];
  // d_in[11] = mask; causal, applied analytically in-kernel.

  float* out  = (float*)d_out;                          // [B,S,D]
  float* attn = (float*)d_out + (size_t)B_ * S_ * D_;   // [B,H,S,S]

  const size_t he = (size_t)B_ * H_ * S_ * DK_;         // elems per projection
  unsigned short* qpb = (unsigned short*)d_ws;          // bf16: 8 MB each
  unsigned short* kpb = qpb + he;
  unsigned short* vpb = kpb + he;
  unsigned short* ctxb = vpb + he;                      // bf16 [B,S,D]; 32 MB total

  dim3 pBlk(128), pGrd(D_ / 64, MROWS / 64);
  proj_f32_bf16hs_kernel<<<pGrd, pBlk, 0, stream>>>(Q, wq, bq, qpb, 0.125f);
  proj_f32_bf16hs_kernel<<<pGrd, pBlk, 0, stream>>>(K, wk, bk, kpb, 1.0f);
  proj_f32_bf16hs_kernel<<<pGrd, pBlk, 0, stream>>>(V, wv, bv, vpb, 1.0f);

  dim3 aBlk(128), aGrd(S_ / 64, H_, B_);
  attention_kernel<<<aGrd, aBlk, 0, stream>>>(qpb, kpb, vpb, attn, ctxb);

  proj_bf16_f32_kernel<<<pGrd, pBlk, 0, stream>>>(ctxb, wo, bo, out);
}
